// MultiViewGNN_48722109006597
// MI455X (gfx1250) — matrix operand order repaired and checked by reading, compiled, not run on hardware
//
#include <hip/hip_runtime.h>
#include <math.h>

#define NNODE 8192
#define FIN   512
#define HD    256

typedef __attribute__((ext_vector_type(16))) _Float16 v16h;
typedef __attribute__((ext_vector_type(8)))  _Float16 v8h;
typedef __attribute__((ext_vector_type(8)))  float    v8f;
typedef unsigned int v4ui __attribute__((ext_vector_type(4)));
typedef int          v8si __attribute__((ext_vector_type(8)));
typedef int          v4si __attribute__((ext_vector_type(4)));

#define WMMA_F16(a,b,c) __builtin_amdgcn_wmma_f32_16x16x32_f16(false,(a),false,(b),(short)0,(c),false,false)

// A operand (16x32 f16): row-major source row; halves [k0+8kg..+7] & [k0+16+8kg..+7]
static __device__ __forceinline__ v16h load_a16(const _Float16* row, int k0, int kg) {
  union { v16h v; v8h h[2]; } u;
  u.h[0] = *(const v8h*)(row + k0 + 8*kg);
  u.h[1] = *(const v8h*)(row + k0 + 16 + 8*kg);
  return u.v;
}
// B operand (32x16 f16): from B^T row (contiguous K); halves [k0+8kg..+7] & [k0+16+8kg..+7]
static __device__ __forceinline__ v16h load_b16(const _Float16* rowT, int k0, int kg) {
  union { v16h v; v8h h[2]; } u;
  u.h[0] = *(const v8h*)(rowT + k0 + 8*kg);
  u.h[1] = *(const v8h*)(rowT + k0 + 16 + 8*kg);
  return u.v;
}

// -------- Tensor Data Mover: 2D tile (f16 elements) global -> LDS --------
static __device__ __forceinline__ unsigned int lds_off(const void* p) {
  return (unsigned int)(uintptr_t)p;   // flat LDS addr truncates to LDS byte offset
}
static __device__ __forceinline__ void tdm_load_2d_f16(
    const void* gaddr, unsigned int ldsaddr,
    unsigned int tensor_d0, unsigned int tensor_d1,
    unsigned int tile_d0, unsigned int tile_d1,
    unsigned int stride0 /* elements */) {
  unsigned long long ga = (unsigned long long)(uintptr_t)gaddr;
  v4ui g0;
  g0[0] = 1u;                                           // count=1, no gather
  g0[1] = ldsaddr;                                      // lds_addr bytes
  g0[2] = (unsigned int)(ga & 0xffffffffull);           // global_addr[31:0]
  g0[3] = (unsigned int)((ga >> 32) & 0x01ffffffull) | (2u << 30); // addr[56:32] | type=2
  v8si g1;
  g1[0] = (int)(1u << 16);                              // data_size=2 bytes
  g1[1] = (int)((tensor_d0 & 0xffffu) << 16);           // tensor_dim0[15:0]
  g1[2] = (int)((tensor_d0 >> 16) | ((tensor_d1 & 0xffffu) << 16));
  g1[3] = (int)((tensor_d1 >> 16) | ((tile_d0 & 0xffffu) << 16));  // tile_dim0
  g1[4] = (int)(tile_d1 & 0xffffu);                     // tile_dim1 (tile_dim2=0)
  g1[5] = (int)stride0;                                 // tensor_dim0_stride[31:0]
  g1[6] = 0;
  g1[7] = 0;
  v4si z4; z4[0] = 0; z4[1] = 0; z4[2] = 0; z4[3] = 0;  // 2D tensor: groups 2/3 unused
  v8si z8; z8[0] = 0; z8[1] = 0; z8[2] = 0; z8[3] = 0;
  z8[4] = 0; z8[5] = 0; z8[6] = 0; z8[7] = 0;
  __builtin_amdgcn_tensor_load_to_lds(g0, g1, z4, z4, z8, 0);
}

// ---------------- utility kernels ----------------
__global__ void zero_f32(float* p, int n) {
  int i = blockIdx.x*blockDim.x + threadIdx.x;
  if (i < n) p[i] = 0.f;
}
__global__ void f32_to_f16(const float* in, _Float16* out, int n) {
  int i = blockIdx.x*blockDim.x + threadIdx.x;
  if (i < n) out[i] = (_Float16)in[i];
}
// W[K][N] (f32) -> WT[N][K] (f16)
__global__ void transpose_to_f16(const float* W, _Float16* WT, int K, int N) {
  int i = blockIdx.x*blockDim.x + threadIdx.x;
  if (i >= K*N) return;
  int k = i / N, n = i % N;
  WT[(size_t)n*K + k] = (_Float16)W[i];
}

// ---------------- GEMM: C(f32)[M][N] = A(f16)[M][K] * B^T(f16)[N][K] ----------------
__global__ void __launch_bounds__(256) gemm_f32out(
    const _Float16* A, const _Float16* BT, float* C, int M, int N, int K) {
  int wave = blockIdx.x * (blockDim.x >> 5) + (threadIdx.x >> 5);
  int lane = threadIdx.x & 31;
  int kg = lane >> 4, hl = lane & 15;
  int ntiles = N >> 4;
  int tm = wave / ntiles, tn = wave % ntiles;
  const _Float16* arow = A + (size_t)(tm*16 + hl) * K;
  const _Float16* brow = BT + (size_t)(tn*16 + hl) * K;
  v8f acc = {};
  for (int k0 = 0; k0 < K; k0 += 32) {
    v16h a = load_a16(arow, k0, kg);
    v16h b = load_b16(brow, k0, kg);
    acc = WMMA_F16(a, b, acc);
  }
  for (int r = 0; r < 8; ++r)
    C[(size_t)(tm*16 + r + 8*kg) * N + tn*16 + hl] = acc[r];
}

// Projection GEMM with bias, f16 output. mode 0: OUT[M][N]; mode 1: OUT[N][M] (transposed)
__global__ void __launch_bounds__(256) gemm_proj(
    const _Float16* A, const _Float16* BT, const float* bias,
    _Float16* OUT, int M, int N, int K, int mode) {
  int wave = blockIdx.x * (blockDim.x >> 5) + (threadIdx.x >> 5);
  int lane = threadIdx.x & 31;
  int kg = lane >> 4, hl = lane & 15;
  int ntiles = N >> 4;
  int tm = wave / ntiles, tn = wave % ntiles;
  const _Float16* arow = A + (size_t)(tm*16 + hl) * K;
  const _Float16* brow = BT + (size_t)(tn*16 + hl) * K;
  v8f acc = {};
  for (int k0 = 0; k0 < K; k0 += 32) {
    v16h a = load_a16(arow, k0, kg);
    v16h b = load_b16(brow, k0, kg);
    acc = WMMA_F16(a, b, acc);
  }
  float bz = bias[tn*16 + hl];
  if (mode == 0) {
    for (int r = 0; r < 8; ++r)
      OUT[(size_t)(tm*16 + r + 8*kg) * N + tn*16 + hl] = (_Float16)(acc[r] + bz);
  } else {
    union { v8h v; _Float16 e[8]; } u;
    for (int r = 0; r < 8; ++r) u.e[r] = (_Float16)(acc[r] + bz);
    *(v8h*)(OUT + (size_t)(tn*16 + hl) * M + tm*16 + 8*kg) = u.v;
  }
}

// ---------------- GCN pieces ----------------
__global__ void deg_init(float* deg, int n) {
  int i = blockIdx.x*blockDim.x + threadIdx.x;
  if (i < n) deg[i] = 1.0f;              // self-loop contributes 1
}
__global__ void deg_acc(const int* ei, const float* ew, float* deg, int E) {
  int e = blockIdx.x*blockDim.x + threadIdx.x;
  if (e < E) atomicAdd(&deg[ei[E + e]], ew[e]);
}
__global__ void dinv_kernel(const float* deg, float* dinv, int n) {
  int i = blockIdx.x*blockDim.x + threadIdx.x;
  if (i < n) { float d = deg[i]; dinv[i] = d > 0.f ? rsqrtf(d) : 0.f; }
}
// per (edge, 8-col segment): agg[dst] += norm * h[src]
__global__ void scatter_edges(const int* ei, const float* ew, const float* dinv,
                              const float* h, float* agg, int E) {
  int g = blockIdx.x*blockDim.x + threadIdx.x;
  int e = g >> 5;
  if (e >= E) return;
  int seg = (g & 31) * 8;
  int src = ei[e], dst = ei[E + e];
  float norm = dinv[src] * ew[e] * dinv[dst];
  const float* hs = h + (size_t)src * HD + seg;
  float* ad = agg + (size_t)dst * HD + seg;
  for (int k = 0; k < 8; ++k) atomicAdd(&ad[k], norm * hs[k]);
}
// feat = relu(agg + dinv^2*h + b); write f16 copy and accumulate into features out
__global__ void feat_kernel(const float* agg, const float* h, const float* dinv,
                            const float* b1, _Float16* fh, float* features,
                            int total, int first) {
  int i = blockIdx.x*blockDim.x + threadIdx.x;
  if (i >= total) return;
  int node = i >> 8, c = i & (HD-1);
  float di = dinv[node];
  float v = agg[i] + di*di*h[i] + b1[c];
  v = fmaxf(v, 0.f);
  fh[i] = (_Float16)v;
  features[i] = first ? v : features[i] + v;
}

// ---------------- Flash attention: S = softmax(Q K^T) V ----------------
// Block = 8 waves = 128 query rows; K/V 32-key tiles staged in LDS by the TDM,
// double-buffered (DMA of block i+1 overlaps compute of block i).
__global__ void __launch_bounds__(256) flash_attn(
    const _Float16* Q, const _Float16* Kh, const _Float16* VT,
    float* S, int Nn) {
  __shared__ _Float16 Kt[2][32][HD];     // 32 keys x 256 dims (row-major, B-op for QK^T)
  __shared__ _Float16 Vt[2][HD][32];     // 256 out-cols x 32 keys (from V^T, B-op for PV)
  __shared__ _Float16 pbuf[8][16*32];    // per-wave P staging
  int wv = threadIdx.x >> 5;
  int lane = threadIdx.x & 31;
  int kg = lane >> 4, hl = lane & 15;
  int qblk = blockIdx.x * 8 + wv;
  const _Float16* qrowp = Q + (size_t)(qblk*16 + hl) * HD;
  v16h qa[8];
  for (int c = 0; c < 8; ++c) qa[c] = load_a16(qrowp, 32*c, kg);
  float mrun[8], lrun[8];
  v8f o[8];
  for (int r = 0; r < 8; ++r) { mrun[r] = -1e30f; lrun[r] = 0.f; }
  for (int c = 0; c < 8; ++c) { v8f z = {}; o[c] = z; }
  _Float16* P = pbuf[wv];

  // prologue: DMA first K/V tiles
  if (wv == 0) {
    tdm_load_2d_f16(Kh, lds_off(&Kt[0][0][0]), HD, (unsigned)Nn, HD, 32, HD);
    tdm_load_2d_f16(VT, lds_off(&Vt[0][0][0]), (unsigned)Nn, HD, 32, HD, (unsigned)Nn);
  }
  int nit = Nn / 32;
  for (int it = 0; it < nit; ++it) {
    int buf = it & 1;
    if (wv == 0) __builtin_amdgcn_s_wait_tensorcnt(0);
    __syncthreads();                      // publish LDS tiles to all waves
    if (wv == 0 && it + 1 < nit) {        // DMA next tiles into other buffer
      int kb2 = (it + 1) * 32;
      tdm_load_2d_f16(Kh + (size_t)kb2 * HD, lds_off(&Kt[buf^1][0][0]),
                      HD, (unsigned)Nn, HD, 32, HD);
      tdm_load_2d_f16(VT + kb2, lds_off(&Vt[buf^1][0][0]),
                      (unsigned)Nn, HD, 32, HD, (unsigned)Nn);
    }
    // scores: 16x32 tile (two 16-key half tiles), f32 accumulate, operands from LDS
    v8f s0 = {}, s1 = {};
    const _Float16* k0p = &Kt[buf][hl][0];
    const _Float16* k1p = &Kt[buf][16 + hl][0];
    for (int c = 0; c < 8; ++c) {
      s0 = WMMA_F16(qa[c], load_b16(k0p, 32*c, kg), s0);
      s1 = WMMA_F16(qa[c], load_b16(k1p, 32*c, kg), s1);
    }
    // online softmax per row r (+8*kg)
    for (int r = 0; r < 8; ++r) {
      float t = fmaxf(s0[r], s1[r]);
      t = fmaxf(t, __shfl_xor(t, 1, 32));
      t = fmaxf(t, __shfl_xor(t, 2, 32));
      t = fmaxf(t, __shfl_xor(t, 4, 32));
      t = fmaxf(t, __shfl_xor(t, 8, 32));
      float mn = fmaxf(mrun[r], t);
      float corr = __expf(mrun[r] - mn);
      float p0 = __expf(s0[r] - mn);
      float p1 = __expf(s1[r] - mn);
      float ls = p0 + p1;
      ls += __shfl_xor(ls, 1, 32);
      ls += __shfl_xor(ls, 2, 32);
      ls += __shfl_xor(ls, 4, 32);
      ls += __shfl_xor(ls, 8, 32);
      lrun[r] = lrun[r]*corr + ls;
      mrun[r] = mn;
      for (int c = 0; c < 8; ++c) o[c][r] *= corr;
      s0[r] = p0; s1[r] = p1;
    }
    // C-layout -> LDS (row-major 16x32 f16), reload as A operand (K=32)
    for (int r = 0; r < 8; ++r) {
      P[(r + 8*kg)*32 + hl]      = (_Float16)s0[r];
      P[(r + 8*kg)*32 + 16 + hl] = (_Float16)s1[r];
    }
    asm volatile("s_wait_dscnt 0" ::: "memory");
    v16h pa;
    { union { v16h v; v8h h[2]; } u;
      u.h[0] = *(const v8h*)(P + hl*32 + 8*kg);
      u.h[1] = *(const v8h*)(P + hl*32 + 16 + 8*kg);
      pa = u.v; }
    // O += P @ Vblock (V tile rows contiguous in key index)
    for (int c = 0; c < 8; ++c) {
      const _Float16* vrow = &Vt[buf][16*c + hl][0];
      o[c] = WMMA_F16(pa, load_b16(vrow, 0, kg), o[c]);
    }
  }
  for (int r = 0; r < 8; ++r) {
    float inv = 1.0f / lrun[r];
    size_t row = (size_t)(qblk*16 + r + 8*kg);
    for (int c = 0; c < 8; ++c)
      S[row * HD + 16*c + hl] = o[c][r] * inv;
  }
}

// ---------------- channel attention ----------------
__global__ void col_reduce(const float* S, float* mx, float* av, int M) {
  int c = threadIdx.x;                          // 256 threads
  float m = -1e30f, s = 0.f;
  for (int r = 0; r < M; ++r) {
    float v = S[(size_t)r * HD + c];
    m = fmaxf(m, v); s += v;
  }
  mx[c] = m; av[c] = s / (float)M;
}
__global__ void chan_mlp(const float* mx, const float* av,
                         const float* Wc1, const float* bc1,
                         const float* Wc2, const float* bc2, float* scale) {
  __shared__ float sm[256], sa[256], hid[32], red[256];
  int t = threadIdx.x;
  sm[t] = mx[t]; sa[t] = av[t];
  __syncthreads();
  if (t < 32) {
    float hm = bc1[t], ha = bc1[t];
    for (int c = 0; c < 256; ++c) { hm += sm[c]*Wc1[c*32+t]; ha += sa[c]*Wc1[c*32+t]; }
    hid[t] = fmaxf(hm, 0.f) + fmaxf(ha, 0.f);
  }
  __syncthreads();
  float v = bc2[t];
  for (int j = 0; j < 32; ++j) v += hid[j]*Wc2[j*256+t];
  red[t] = v; __syncthreads();
  for (int s2 = 128; s2 > 0; s2 >>= 1) { if (t < s2) red[t] = fmaxf(red[t], red[t+s2]); __syncthreads(); }
  float mxv = red[0]; __syncthreads();
  float e = __expf(v - mxv);
  red[t] = e; __syncthreads();
  for (int s2 = 128; s2 > 0; s2 >>= 1) { if (t < s2) red[t] += red[t+s2]; __syncthreads(); }
  scale[t] = e / red[0];
}
__global__ void xmv_acc(const float* S, const float* scale, float* xmv,
                        int total, int first) {
  int i = blockIdx.x*blockDim.x + threadIdx.x;
  if (i >= total) return;
  float v = S[i] * scale[i & (HD-1)];
  xmv[i] = first ? v : xmv[i] + v;
}
// out[i] = relu(dot(xmv[i,:], Wl) + bl)   (one wave per node)
__global__ void out_kernel(const float* xmv, const float* Wl, const float* bl,
                           float* out, int N) {
  int wv = blockIdx.x * (blockDim.x >> 5) + (threadIdx.x >> 5);
  int lane = threadIdx.x & 31;
  if (wv >= N) return;
  float s = 0.f;
  for (int c = lane; c < HD; c += 32) s += xmv[(size_t)wv*HD + c] * Wl[c];
  s += __shfl_xor(s, 1, 32);
  s += __shfl_xor(s, 2, 32);
  s += __shfl_xor(s, 4, 32);
  s += __shfl_xor(s, 8, 32);
  s += __shfl_xor(s, 16, 32);
  if (lane == 0) out[wv] = fmaxf(s + bl[0], 0.f);
}

// ---------------- host ----------------
extern "C" void kernel_launch(void* const* d_in, const int* in_sizes, int n_in,
                              void* d_out, int out_size, void* d_ws, size_t ws_size,
                              hipStream_t stream) {
  const float* x = (const float*)d_in[0];
  const int*   ei[3] = {(const int*)d_in[1], (const int*)d_in[5], (const int*)d_in[9]};
  const float* ew[3] = {(const float*)d_in[2], (const float*)d_in[6], (const float*)d_in[10]};
  const float* W1[3] = {(const float*)d_in[3], (const float*)d_in[7], (const float*)d_in[11]};
  const float* b1[3] = {(const float*)d_in[4], (const float*)d_in[8], (const float*)d_in[12]};
  const float* Wq = (const float*)d_in[13]; const float* bq = (const float*)d_in[14];
  const float* Wk = (const float*)d_in[15]; const float* bk = (const float*)d_in[16];
  const float* Wv = (const float*)d_in[17]; const float* bv = (const float*)d_in[18];
  const float* Wc1 = (const float*)d_in[19]; const float* bc1 = (const float*)d_in[20];
  const float* Wc2 = (const float*)d_in[21]; const float* bc2 = (const float*)d_in[22];
  const float* Wl = (const float*)d_in[23]; const float* bl = (const float*)d_in[24];
  const int E = in_sizes[2];

  // workspace layout
  size_t off = 0;
  char* base = (char*)d_ws;
  auto alloc = [&](size_t bytes) -> char* {
    char* p = base + off; off += (bytes + 255) & ~(size_t)255; return p;
  };
  _Float16* xh  = (_Float16*)alloc((size_t)NNODE*FIN*2);
  _Float16* wqT = (_Float16*)alloc((size_t)HD*HD*2);
  _Float16* wkT = (_Float16*)alloc((size_t)HD*HD*2);
  _Float16* wvT = (_Float16*)alloc((size_t)HD*HD*2);
  _Float16* w1t = (_Float16*)alloc((size_t)HD*FIN*2);
  float*    h   = (float*)   alloc((size_t)NNODE*HD*4);
  float*    agg = (float*)   alloc((size_t)NNODE*HD*4);
  _Float16* fh  = (_Float16*)alloc((size_t)NNODE*HD*2);
  _Float16* qh  = (_Float16*)alloc((size_t)NNODE*HD*2);
  _Float16* kh  = (_Float16*)alloc((size_t)NNODE*HD*2);
  _Float16* vT  = (_Float16*)alloc((size_t)NNODE*HD*2);
  float*    sbuf= (float*)   alloc((size_t)NNODE*HD*4);
  float*    xmv = (float*)   alloc((size_t)NNODE*HD*4);
  float*    deg = (float*)   alloc((size_t)NNODE*4);
  float*    dinv= (float*)   alloc((size_t)NNODE*4);
  float*    mx  = (float*)   alloc(1024);
  float*    av  = (float*)   alloc(1024);
  float*    scl = (float*)   alloc(1024);

  float* out_vec  = (float*)d_out;          // [NNODE]
  float* features = (float*)d_out + NNODE;  // [NNODE*HD]

  const int nhd = NNODE * HD;               // 2097152

  // shared preprocessing
  f32_to_f16<<<(NNODE*FIN)/256, 256, 0, stream>>>(x, xh, NNODE*FIN);
  transpose_to_f16<<<(HD*HD)/256, 256, 0, stream>>>(Wq, wqT, HD, HD);
  transpose_to_f16<<<(HD*HD)/256, 256, 0, stream>>>(Wk, wkT, HD, HD);
  transpose_to_f16<<<(HD*HD)/256, 256, 0, stream>>>(Wv, wvT, HD, HD);

  for (int v = 0; v < 3; ++v) {
    // ---- GCN ----
    transpose_to_f16<<<(FIN*HD)/256, 256, 0, stream>>>(W1[v], w1t, FIN, HD);
    gemm_f32out<<<(NNODE/16)*(HD/16)/8, 256, 0, stream>>>(xh, w1t, h, NNODE, HD, FIN);
    deg_init<<<NNODE/256, 256, 0, stream>>>(deg, NNODE);
    deg_acc<<<E/256, 256, 0, stream>>>(ei[v], ew[v], deg, E);
    dinv_kernel<<<NNODE/256, 256, 0, stream>>>(deg, dinv, NNODE);
    zero_f32<<<nhd/256, 256, 0, stream>>>(agg, nhd);
    scatter_edges<<<(E*32)/256, 256, 0, stream>>>(ei[v], ew[v], dinv, h, agg, E);
    feat_kernel<<<nhd/256, 256, 0, stream>>>(agg, h, dinv, b1[v], fh, features, nhd, v == 0);
    // ---- spatial attention ----
    gemm_proj<<<(NNODE/16)*(HD/16)/8, 256, 0, stream>>>(fh, wqT, bq, qh, NNODE, HD, HD, 0);
    gemm_proj<<<(NNODE/16)*(HD/16)/8, 256, 0, stream>>>(fh, wkT, bk, kh, NNODE, HD, HD, 0);
    gemm_proj<<<(NNODE/16)*(HD/16)/8, 256, 0, stream>>>(fh, wvT, bv, vT, NNODE, HD, HD, 1);
    flash_attn<<<NNODE/16/8, 256, 0, stream>>>(qh, kh, vT, sbuf, NNODE);
    // ---- channel attention ----
    col_reduce<<<1, 256, 0, stream>>>(sbuf, mx, av, NNODE);
    chan_mlp<<<1, 256, 0, stream>>>(mx, av, Wc1, bc1, Wc2, bc2, scl);
    xmv_acc<<<nhd/256, 256, 0, stream>>>(sbuf, scl, xmv, nhd, v == 0);
  }
  out_kernel<<<NNODE/8, 256, 0, stream>>>(xmv, Wl, bl, out_vec, NNODE);
}
